// Decoder_7481833030033
// MI455X (gfx1250) — compile-verified, matrix operand order checked by reading, not run on hardware
//
#include <hip/hip_runtime.h>
#include <hip/hip_bf16.h>
#include <math.h>

// Problem constants
#define NB   256
#define NP   196
#define NENC 2048
#define NV   10000
#define NE   512
#define NH   512
#define NS   54
#define NT   53

typedef __attribute__((ext_vector_type(16))) __bf16 v16bf;
typedef __attribute__((ext_vector_type(8)))  float  v8f;
typedef __hip_bfloat16 bf16;

#if defined(__has_builtin)
#if __has_builtin(__builtin_amdgcn_global_load_async_to_lds_b128) && \
    __has_builtin(__builtin_amdgcn_s_wait_asynccnt)
#define HAS_ASYNC 1
#endif
#endif

#ifdef HAS_ASYNC
typedef int v4i __attribute__((vector_size(16)));
typedef __attribute__((address_space(1))) v4i* gp4;   // global int4*
typedef __attribute__((address_space(3))) v4i* lp4;   // LDS int4*
#endif

union FragU { unsigned u[8]; v16bf v; };
struct TileCtx { v8f acc[2][4]; int row0, col0; };

// LDS tile row stride in dwords: 20 (80B rows: 16B-aligned for b128 stores,
// and 20*m mod 64 distinct for m=0..15 -> conflict-free fragment reads)
#define LSTR 20

// ---------------------------------------------------------------------------
// WMMA GEMM core:  C[M,N] = A[M,K] * W[N,K]^T   (A,W bf16 row-major, K-contig)
// 256 thr = 8 wave32 (4 M x 2 N), block tile 128x128, wave tile 32x64.
// Double-buffered LDS; async memory->LDS staging one K-chunk ahead.
// M multiple of 128; N-edge handled by clamped source rows (OOB columns are
// never stored by the epilogue); K multiple of 32.
// ---------------------------------------------------------------------------
__device__ __forceinline__ void gemm_core(const bf16* __restrict__ A,
                                          const bf16* __restrict__ W,
                                          int N, int K, TileCtx& tc)
{
    __shared__ unsigned sA[2][128 * LSTR];
    __shared__ unsigned sB[2][128 * LSTR];
    const int tid  = threadIdx.x;
    const int lane = tid & 31;
    const int wv   = tid >> 5;
    const int wm   = wv >> 1, wn = wv & 1;
    const int m_blk = blockIdx.y * 128;
    const int n_blk = blockIdx.x * 128;
    tc.row0 = m_blk + wm * 32;
    tc.col0 = n_blk + wn * 64;

    v8f zero;
    for (int i = 0; i < 8; i++) zero[i] = 0.0f;
    for (int i = 0; i < 2; i++)
        for (int j = 0; j < 4; j++) tc.acc[i][j] = zero;

    const int g  = lane >> 4;
    const int mr = lane & 15;
    const int Kd = K >> 1;                 // K in dwords
    const unsigned* Ad = (const unsigned*)A;
    const unsigned* Wd = (const unsigned*)W;

    // staging coords: tile = 128 rows x 16 dwords = 512 uint4, 2 per thread
    const int r0 = tid >> 2;               // 0..63
    const int r1 = r0 + 64;                // 64..127
    const int q  = (tid & 3) * 4;          // dword col 0/4/8/12
    // clamp W rows at the N edge: duplicated rows only feed masked-out columns
    const int nr0 = min(n_blk + r0, N - 1);
    const int nr1 = min(n_blk + r1, N - 1);
    const int nk = K >> 5;

    // stage one 32-K chunk into LDS buffer `buf` (4 async b128 ops / thread)
    auto issue = [&](int buf, int kd0) {
        unsigned* a0d = &sA[buf][r0 * LSTR + q];
        unsigned* a1d = &sA[buf][r1 * LSTR + q];
        unsigned* b0d = &sB[buf][r0 * LSTR + q];
        unsigned* b1d = &sB[buf][r1 * LSTR + q];
        const unsigned* a0s = Ad + (size_t)(m_blk + r0) * Kd + kd0 + q;
        const unsigned* a1s = Ad + (size_t)(m_blk + r1) * Kd + kd0 + q;
        const unsigned* b0s = Wd + (size_t)nr0 * Kd + kd0 + q;
        const unsigned* b1s = Wd + (size_t)nr1 * Kd + kd0 + q;
#ifdef HAS_ASYNC
        __builtin_amdgcn_global_load_async_to_lds_b128((gp4)(v4i*)a0s, (lp4)(v4i*)a0d, 0, 0);
        __builtin_amdgcn_global_load_async_to_lds_b128((gp4)(v4i*)a1s, (lp4)(v4i*)a1d, 0, 0);
        __builtin_amdgcn_global_load_async_to_lds_b128((gp4)(v4i*)b0s, (lp4)(v4i*)b0d, 0, 0);
        __builtin_amdgcn_global_load_async_to_lds_b128((gp4)(v4i*)b1s, (lp4)(v4i*)b1d, 0, 0);
#else
        uint4 va0 = *(const uint4*)a0s;
        uint4 va1 = *(const uint4*)a1s;
        uint4 vb0 = *(const uint4*)b0s;
        uint4 vb1 = *(const uint4*)b1s;
        *(uint4*)a0d = va0;
        *(uint4*)a1d = va1;
        *(uint4*)b0d = vb0;
        *(uint4*)b1d = vb1;
#endif
    };

    issue(0, 0);                            // prologue: chunk 0 -> buf 0
    for (int kc = 0; kc < nk; kc++) {
        const int cur = kc & 1;
#ifdef HAS_ASYNC
        __builtin_amdgcn_s_wait_asynccnt(0);  // own async writes to `cur` done
#endif
        __syncthreads();                    // everyone's writes done, prev reads done
        if (kc + 1 < nk) issue(cur ^ 1, (kc + 1) * 16);  // overlaps with WMMA below

        FragU fa[2], fb[4];
#pragma unroll
        for (int v = 0; v < 8; v++) {
            int cd = ((v < 4) ? 0 : 8) + g * 4 + (v & 3);
            fa[0].u[v] = sA[cur][(wm * 32 +      mr) * LSTR + cd];
            fa[1].u[v] = sA[cur][(wm * 32 + 16 + mr) * LSTR + cd];
#pragma unroll
            for (int j = 0; j < 4; j++)
                fb[j].u[v] = sB[cur][(wn * 64 + j * 16 + mr) * LSTR + cd];
        }
#pragma unroll
        for (int i = 0; i < 2; i++)
#pragma unroll
            for (int j = 0; j < 4; j++)
                tc.acc[i][j] = __builtin_amdgcn_wmma_f32_16x16x32_bf16(
                    false, fa[i].v, false, fb[j].v, (short)0, tc.acc[i][j],
                    false, false);
    }
}

// Epilogue: element (i,j,r,lane): row = row0+i*16+8*(lane>>4)+r,
// col = col0+j*16+(lane&15)   (32-bit C/D layout, ISA 7.12.2)
#define EPI_BEGIN()                                               \
    const int lane_ = threadIdx.x & 31;                           \
    const int lg_ = lane_ >> 4, ln_ = lane_ & 15;                 \
    for (int i_ = 0; i_ < 2; i_++)                                \
    for (int j_ = 0; j_ < 4; j_++)                                \
    for (int r_ = 0; r_ < 8; r_++) {                              \
        int row = tc.row0 + i_ * 16 + lg_ * 8 + r_;               \
        int col = tc.col0 + j_ * 16 + ln_;                        \
        float val = tc.acc[i_][j_][r_];
#define EPI_END() }

__global__ void k_gemm_f32(const bf16* A, const bf16* W, const float* bias,
                           float* out, int N, int K) {
    TileCtx tc; gemm_core(A, W, N, K, tc);
    EPI_BEGIN()
        if (col < N) out[(size_t)row * N + col] = val + bias[col];
    EPI_END()
}

__global__ void k_gemm_b16(const bf16* A, const bf16* W, const float* bias,
                           bf16* out, int N, int K) {
    TileCtx tc; gemm_core(A, W, N, K, tc);
    EPI_BEGIN()
        if (col < N) out[(size_t)row * N + col] = __float2bfloat16(val + bias[col]);
    EPI_END()
}

// hid_att (cols 0:512) + sigmoid sag gate (cols 512:2560)
__global__ void k_gemm_hs(const bf16* A, const bf16* W, const float* bias,
                          float* hid, float* gs, int N, int K) {
    TileCtx tc; gemm_core(A, W, N, K, tc);
    EPI_BEGIN()
        float v = val + bias[col];
        if (col < NH) hid[row * NH + col] = v;
        else          gs[row * NENC + (col - NH)] = 1.0f / (1.0f + expf(-v));
    EPI_END()
}

// h0 (cols 0:512, also bf16 copy) and c0 (cols 512:1024)
__global__ void k_gemm_init(const bf16* A, const bf16* W, const float* bias,
                            float* h, float* c, bf16* h_bf, int N, int K) {
    TileCtx tc; gemm_core(A, W, N, K, tc);
    EPI_BEGIN()
        float v = val + bias[col];
        if (col < NH) { h[row * NH + col] = v; h_bf[row * NH + col] = __float2bfloat16(v); }
        else            c[row * NH + (col - NH)] = v;
    EPI_END()
}

// final logits: A rows are [t*256 + b]; mask inactive (t >= pred_len[b]) to 0
__global__ void k_gemm_logits(const bf16* A, const bf16* W, const float* bias,
                              const int* pred_len, float* preds, int N, int K) {
    TileCtx tc; gemm_core(A, W, N, K, tc);
    EPI_BEGIN()
        if (col < N) {
            int t = row >> 8, b = row & 255;
            float v = (t < pred_len[b]) ? (val + bias[col]) : 0.0f;
            preds[((size_t)b * NT + t) * N + col] = v;
        }
    EPI_END()
}

// ---------------------------------------------------------------------------
// Pointwise / prep kernels
// ---------------------------------------------------------------------------
__global__ void k_order(const int* lens, const int* caps, int* order,
                        int* pred_len_i, int* caps_s,
                        float* out_predlen, float* out_caps) {
    __shared__ int s_ord[NB];
    int tid = threadIdx.x;
    int lt = lens[tid];
    int rank = 0;
    for (int j = 0; j < NB; j++) {
        int lj = lens[j];
        rank += (lj > lt) || (lj == lt && j < tid);   // stable argsort(-lens)
    }
    s_ord[rank] = tid;
    __syncthreads();
    int src = s_ord[tid];
    order[tid] = src;
    int pl = lens[src] - 1;
    pred_len_i[tid] = pl;
    out_predlen[tid] = (float)pl;
    for (int s = 0; s < NS; s++) {
        int cv = caps[src * NS + s];
        caps_s[tid * NS + s] = cv;
        out_caps[tid * NS + s] = (float)cv;
    }
}

__global__ void k_conv_enc(const float* enc, const int* order, bf16* out, size_t n) {
    const size_t per = (size_t)NP * NENC;
    for (size_t i = (size_t)blockIdx.x * blockDim.x + threadIdx.x; i < n;
         i += (size_t)gridDim.x * blockDim.x) {
        size_t b = i / per, rem = i % per;
        out[i] = __float2bfloat16(enc[(size_t)order[b] * per + rem]);
    }
}

__global__ void k_mean(const float* enc, const int* order, float* mean, bf16* mean_bf) {
    int b = blockIdx.y;
    int ch = blockIdx.x * 256 + threadIdx.x;
    const float* e = enc + (size_t)order[b] * NP * NENC + ch;
    float s = 0.f;
    for (int p = 0; p < NP; p++) s += e[(size_t)p * NENC];
    float v = s * (1.0f / (float)NP);
    mean[b * NENC + ch] = v;
    mean_bf[b * NENC + ch] = __float2bfloat16(v);
}

__global__ void k_f2b(const float* in, bf16* out, size_t n) {
    for (size_t i = (size_t)blockIdx.x * blockDim.x + threadIdx.x; i < n;
         i += (size_t)gridDim.x * blockDim.x)
        out[i] = __float2bfloat16(in[i]);
}

__global__ void k_cat_rows(const float* A, int rowsA, const float* Bm, int rowsB,
                           int K, bf16* out) {
    size_t n = (size_t)(rowsA + rowsB) * K;
    for (size_t i = (size_t)blockIdx.x * blockDim.x + threadIdx.x; i < n;
         i += (size_t)gridDim.x * blockDim.x) {
        int row = (int)(i / K), col = (int)(i % K);
        float v = (row < rowsA) ? A[(size_t)row * K + col]
                                : Bm[(size_t)(row - rowsA) * K + col];
        out[i] = __float2bfloat16(v);
    }
}

__global__ void k_cat_cols(const float* A, int colsA, const float* Bm, int colsB,
                           int rows, bf16* out) {
    int K = colsA + colsB;
    size_t n = (size_t)rows * K;
    for (size_t i = (size_t)blockIdx.x * blockDim.x + threadIdx.x; i < n;
         i += (size_t)gridDim.x * blockDim.x) {
        int row = (int)(i / K), col = (int)(i % K);
        float v = (col < colsA) ? A[(size_t)row * colsA + col]
                                : Bm[(size_t)row * colsB + (col - colsA)];
        out[i] = __float2bfloat16(v);
    }
}

__global__ void k_cat_vec(const float* a, int na, const float* b, int nb, float* out) {
    int i = blockIdx.x * 256 + threadIdx.x;
    if (i < na + nb) out[i] = (i < na) ? a[i] : b[i - na];
}

__global__ void k_add_vec(const float* a, const float* b, float* out, int n) {
    int i = blockIdx.x * 256 + threadIdx.x;
    if (i < n) out[i] = a[i] + b[i];
}

__global__ void k_emb(const float* emb, const int* caps_s, bf16* emb_x, size_t n) {
    for (size_t i = (size_t)blockIdx.x * blockDim.x + threadIdx.x; i < n;
         i += (size_t)gridDim.x * blockDim.x) {
        size_t b = i / ((size_t)NT * NE);
        size_t r = i % ((size_t)NT * NE);
        int t = (int)(r / NE), e = (int)(r % NE);
        int tok = caps_s[b * NS + t];
        emb_x[i] = __float2bfloat16(emb[(size_t)tok * NE + e]);
    }
}

__global__ void k_zero(float* p, size_t n) {
    for (size_t i = (size_t)blockIdx.x * blockDim.x + threadIdx.x; i < n;
         i += (size_t)gridDim.x * blockDim.x)
        p[i] = 0.0f;
}

// attention: scores + softmax + gated context; assemble A-row for gates GEMM
__global__ void k_attn(const bf16* enc_att, const float* hid_att, const bf16* enc_s,
                       const float* gs, const bf16* emb_x, const bf16* h_bf,
                       const float* att_f_w, const float* att_f_b,
                       bf16* a_step, int t) {
    const int b = blockIdx.x;
    const int tid = threadIdx.x;
    const int lane = tid & 31, wv = tid >> 5;
    __shared__ float s_e[NP];
    __shared__ float s_red[8];

    const bf16*  ea = enc_att + (size_t)b * NP * NH;
    const float* ha = hid_att + b * NH;
    for (int p = wv; p < NP; p += 8) {
        float acc = 0.f;
        for (int k = lane; k < NH; k += 32) {
            float v = __bfloat162float(ea[p * NH + k]) + ha[k];
            v = v > 0.f ? v : 0.f;
            acc += v * att_f_w[k];
        }
        for (int off = 16; off; off >>= 1) acc += __shfl_down(acc, off, 32);
        if (lane == 0) s_e[p] = acc + att_f_b[0];
    }
    __syncthreads();
    float m = -1e30f;
    for (int p = tid; p < NP; p += 256) m = fmaxf(m, s_e[p]);
    for (int off = 16; off; off >>= 1) m = fmaxf(m, __shfl_down(m, off, 32));
    if (lane == 0) s_red[wv] = m;
    __syncthreads();
    m = s_red[0];
    for (int i = 1; i < 8; i++) m = fmaxf(m, s_red[i]);
    float sum = 0.f;
    for (int p = tid; p < NP; p += 256) { float e = expf(s_e[p] - m); s_e[p] = e; sum += e; }
    for (int off = 16; off; off >>= 1) sum += __shfl_down(sum, off, 32);
    if (lane == 0) s_red[wv] = sum;
    __syncthreads();
    sum = 0.f;
    for (int i = 0; i < 8; i++) sum += s_red[i];
    float inv = 1.0f / sum;

    bf16* ao = a_step + (size_t)b * 3072;
    const bf16*  es  = enc_s + (size_t)b * NP * NENC;
    const float* gsb = gs + b * NENC;
    for (int ch = tid; ch < NENC; ch += 256) {
        float acc = 0.f;
        for (int p = 0; p < NP; p++)
            acc += s_e[p] * __bfloat162float(es[(size_t)p * NENC + ch]);
        ao[NE + ch] = __float2bfloat16(acc * inv * gsb[ch]);
    }
    const bf16* xb = emb_x + ((size_t)b * NT + t) * NE;
    const bf16* hb = h_bf + b * NH;
    for (int e = tid; e < NE; e += 256) {
        ao[e] = xb[e];            // x_t
        ao[2560 + e] = hb[e];     // h (for W_hh part)
    }
}

__global__ void k_lstm(const float* gates, float* h, float* c, bf16* h_bf,
                       bf16* h_seq, const int* pred_len, int t) {
    int idx = blockIdx.x * 256 + threadIdx.x;   // b*512 + j
    int b = idx >> 9, j = idx & 511;
    const float* gb = gates + (size_t)b * 2048;
    float ig = 1.f / (1.f + expf(-gb[j]));
    float fg = 1.f / (1.f + expf(-gb[512 + j]));
    float gg = tanhf(gb[1024 + j]);
    float og = 1.f / (1.f + expf(-gb[1536 + j]));
    float c_old = c[idx];
    float cn = fg * c_old + ig * gg;
    float hn = og * tanhf(cn);
    bool active = t < pred_len[b];
    float hc = active ? hn : h[idx];
    float cc = active ? cn : c_old;
    h[idx] = hc;
    c[idx] = cc;
    h_bf[idx] = __float2bfloat16(hc);
    h_seq[((size_t)t * NB + b) * NH + j] = __float2bfloat16(hn);
}

// ---------------------------------------------------------------------------
extern "C" void kernel_launch(void* const* d_in, const int* in_sizes, int n_in,
                              void* d_out, int out_size, void* d_ws, size_t ws_size,
                              hipStream_t stream) {
    (void)in_sizes; (void)n_in; (void)out_size; (void)ws_size;
    const float* enc     = (const float*)d_in[0];
    const int*   caps    = (const int*)d_in[1];
    const int*   lens    = (const int*)d_in[2];
    const float* emb     = (const float*)d_in[3];
    const float* W_ih    = (const float*)d_in[4];
    const float* W_hh    = (const float*)d_in[5];
    const float* b_ih    = (const float*)d_in[6];
    const float* b_hh    = (const float*)d_in[7];
    const float* ec_w    = (const float*)d_in[8];
    const float* ec_b    = (const float*)d_in[9];
    const float* eh_w    = (const float*)d_in[10];
    const float* eh_b    = (const float*)d_in[11];
    const float* sag_w   = (const float*)d_in[12];
    const float* sag_b   = (const float*)d_in[13];
    const float* att_e_w = (const float*)d_in[14];
    const float* att_e_b = (const float*)d_in[15];
    const float* att_h_w = (const float*)d_in[16];
    const float* att_h_b = (const float*)d_in[17];
    const float* att_f_w = (const float*)d_in[18];
    const float* att_f_b = (const float*)d_in[19];
    const float* fc_w    = (const float*)d_in[20];
    const float* fc_b    = (const float*)d_in[21];

    float* out         = (float*)d_out;
    float* out_predlen = out;
    float* out_caps    = out + NB;
    float* out_preds   = out + NB + NB * NS;
    float* out_coefs   = out_preds + (size_t)NB * NT * NV;

    char* ws = (char*)d_ws;
    size_t off = 0;
    auto alloc = [&](size_t bytes) -> char* {
        off = (off + 255) & ~(size_t)255;
        char* p = ws + off;
        off += bytes;
        return p;
    };
    int*   order      = (int*)alloc(NB * 4);
    int*   pred_len_i = (int*)alloc(NB * 4);
    int*   caps_s     = (int*)alloc((size_t)NB * NS * 4);
    bf16*  enc_s      = (bf16*)alloc((size_t)NB * NP * NENC * 2);
    float* mean       = (float*)alloc((size_t)NB * NENC * 4);
    bf16*  mean_bf    = (bf16*)alloc((size_t)NB * NENC * 2);
    bf16*  enc_att    = (bf16*)alloc((size_t)NB * NP * NH * 2);
    bf16*  att_e_wb   = (bf16*)alloc((size_t)NH * NENC * 2);
    bf16*  fc_wb      = (bf16*)alloc((size_t)NV * NH * 2);
    bf16*  W_cat      = (bf16*)alloc((size_t)2048 * 3072 * 2);
    bf16*  W_hs       = (bf16*)alloc((size_t)2560 * 512 * 2);
    bf16*  W_init     = (bf16*)alloc((size_t)1024 * 2048 * 2);
    float* bias_hs    = (float*)alloc(2560 * 4);
    float* bias_init  = (float*)alloc(1024 * 4);
    float* bias_g     = (float*)alloc(2048 * 4);
    bf16*  emb_x      = (bf16*)alloc((size_t)NB * NT * NE * 2);
    float* h          = (float*)alloc((size_t)NB * NH * 4);
    float* c          = (float*)alloc((size_t)NB * NH * 4);
    bf16*  h_bf       = (bf16*)alloc((size_t)NB * NH * 2);
    float* hid_att    = (float*)alloc((size_t)NB * NH * 4);
    float* gsb        = (float*)alloc((size_t)NB * NENC * 4);
    bf16*  a_step     = (bf16*)alloc((size_t)NB * 3072 * 2);
    float* gates      = (float*)alloc((size_t)NB * 2048 * 4);
    bf16*  h_seq      = (bf16*)alloc((size_t)NT * NB * NH * 2);

    // prep
    k_order<<<1, 256, 0, stream>>>(lens, caps, order, pred_len_i, caps_s,
                                   out_predlen, out_caps);
    k_conv_enc<<<4096, 256, 0, stream>>>(enc, order, enc_s, (size_t)NB * NP * NENC);
    k_mean<<<dim3(NENC / 256, NB), 256, 0, stream>>>(enc, order, mean, mean_bf);
    k_f2b<<<2048, 256, 0, stream>>>(att_e_w, att_e_wb, (size_t)NH * NENC);
    k_f2b<<<2048, 256, 0, stream>>>(fc_w, fc_wb, (size_t)NV * NH);
    k_cat_cols<<<4096, 256, 0, stream>>>(W_ih, 2560, W_hh, 512, 2048, W_cat);
    k_cat_rows<<<2048, 256, 0, stream>>>(att_h_w, 512, sag_w, 2048, 512, W_hs);
    k_cat_rows<<<2048, 256, 0, stream>>>(eh_w, 512, ec_w, 512, 2048, W_init);
    k_cat_vec<<<10, 256, 0, stream>>>(att_h_b, 512, sag_b, 2048, bias_hs);
    k_cat_vec<<<4, 256, 0, stream>>>(eh_b, 512, ec_b, 512, bias_init);
    k_add_vec<<<8, 256, 0, stream>>>(b_ih, b_hh, bias_g, 2048);
    k_emb<<<4096, 256, 0, stream>>>(emb, caps_s, emb_x, (size_t)NB * NT * NE);
    k_zero<<<2048, 256, 0, stream>>>(out_coefs, (size_t)NB * NT * NP);

    // h0/c0: (256 x 2048) @ (1024 x 2048)^T
    k_gemm_init<<<dim3(1024 / 128, NB / 128), 256, 0, stream>>>(
        mean_bf, W_init, bias_init, h, c, h_bf, 1024, 2048);
    // enc_att: (50176 x 2048) @ (512 x 2048)^T
    k_gemm_b16<<<dim3(NH / 128, (NB * NP) / 128), 256, 0, stream>>>(
        enc_s, att_e_wb, att_e_b, enc_att, NH, NENC);

    // sequential decode
    for (int t = 0; t < NT; t++) {
        k_gemm_hs<<<dim3(2560 / 128, NB / 128), 256, 0, stream>>>(
            h_bf, W_hs, bias_hs, hid_att, gsb, 2560, 512);
        k_attn<<<NB, 256, 0, stream>>>(enc_att, hid_att, enc_s, gsb, emb_x,
                                       h_bf, att_f_w, att_f_b, a_step, t);
        k_gemm_f32<<<dim3(2048 / 128, NB / 128), 256, 0, stream>>>(
            a_step, W_cat, bias_g, gates, 2048, 3072);
        k_lstm<<<(NB * NH) / 256, 256, 0, stream>>>(gates, h, c, h_bf, h_seq,
                                                    pred_len_i, t);
    }

    // logits: (13568 x 512) @ (10000 x 512)^T, masked scatter into preds
    k_gemm_logits<<<dim3((NV + 127) / 128, (NT * NB) / 128), 256, 0, stream>>>(
        h_seq, fc_wb, fc_b, pred_len_i, out_preds, NV, NH);
}
